// MultiHeadAttention_37160057045042
// MI455X (gfx1250) — matrix operand order repaired and checked by reading, not compiled, not run on hardware
//
#include <hip/hip_runtime.h>
#include <math.h>

// ---------------------------------------------------------------------------
// MHA pipeline for gfx1250 (MI455X): wave32, WMMA f32_16x16x32_f16,
// TDM (tensor_load_to_lds) GEMM staging with double buffering, async
// global->LDS staging in flash-attention.
//   B=2, S=2048, D=1024, H=16, DK=64.
// ---------------------------------------------------------------------------

typedef __attribute__((ext_vector_type(16))) _Float16     v16h;
typedef __attribute__((ext_vector_type(8)))  float        v8f;
typedef __attribute__((ext_vector_type(4)))  unsigned int u32x4;
typedef __attribute__((ext_vector_type(8)))  int          i32x8;
typedef __attribute__((ext_vector_type(4)))  int          i32x4;

#if __has_builtin(__builtin_amdgcn_tensor_load_to_lds)
#define HAVE_TDM 1
#else
#define HAVE_TDM 0
#endif

union FragU { v16h v; uint4 u[2]; };

__device__ __forceinline__ v8f zero8() {
    v8f z = {0.f, 0.f, 0.f, 0.f, 0.f, 0.f, 0.f, 0.f};
    return z;
}

// Async copy of 16 bytes from global (SGPR base + 32-bit byte offset) straight
// into LDS at byte address `lds_addr`. Tracked by ASYNCcnt.
__device__ __forceinline__ void async_load_b128(unsigned int lds_addr,
                                                const void* gbase,
                                                unsigned int byte_off) {
    asm volatile("global_load_async_to_lds_b128 %0, %1, %2"
                 :
                 : "v"(lds_addr), "v"(byte_off), "s"(gbase)
                 : "memory");
}

__device__ __forceinline__ void wait_async0() {
    asm volatile("s_wait_asynccnt 0" ::: "memory");
}

__device__ __forceinline__ unsigned int lds_addr_of(const void* p) {
    return (unsigned int)(unsigned long long)p;  // flat LDS addr: low 32 bits
}

#if HAVE_TDM
// One TDM 2D tile load: tile_y rows x tile_x elements (f16), row pitch
// `stride_elems`, packed contiguously into LDS at `lds_addr`.
// D# layout per CDNA5 ISA ch.8 (group0 128b, group1 256b).
__device__ __forceinline__ void tdm_load_2d(unsigned int lds_addr,
                                            const void* gaddr,
                                            unsigned int tile_x,
                                            unsigned int tile_y,
                                            unsigned int stride_elems) {
    unsigned long long ga = (unsigned long long)gaddr;
    u32x4 g0;
    g0[0] = 1u;                                   // count=1, user descriptor
    g0[1] = lds_addr;                             // LDS byte address
    g0[2] = (unsigned int)ga;                     // global_addr[31:0]
    g0[3] = (unsigned int)((ga >> 32) & 0x1FFFFFFull) | (2u << 30);  // type=2
    i32x8 g1;
    g1[0] = (int)(1u << 16);                      // data_size=1 (2 bytes)
    g1[1] = (int)((tile_x & 0xFFFFu) << 16);      // tensor_dim0 = tile_x
    g1[2] = (int)((tile_x >> 16) | ((tile_y & 0xFFFFu) << 16)); // tensor_dim1
    g1[3] = (int)((tile_y >> 16) | ((tile_x & 0xFFFFu) << 16)); // tile_dim0
    g1[4] = (int)(tile_y & 0xFFFFu);              // tile_dim1 (tile_dim2=0)
    g1[5] = (int)stride_elems;                    // tensor_dim0_stride[31:0]
    g1[6] = 0;                                    // stride[47:32], dim1_stride
    g1[7] = 0;
    i32x4 gz = {0, 0, 0, 0};
#if defined(__clang_major__) && __clang_major__ >= 23
    i32x8 gz8 = {0, 0, 0, 0, 0, 0, 0, 0};
    __builtin_amdgcn_tensor_load_to_lds(g0, g1, gz, gz, gz8, 0);
#else
    __builtin_amdgcn_tensor_load_to_lds(g0, g1, gz, gz, 0);
#endif
}
#endif

// A-matrix fragment (16x32 f16, MxK): lanes 0-15 hold rows M=lane with
// K = {0..7, 16..23}; lanes 16-31 hold rows M=lane-16 with K = {8..15, 24..31}.
__device__ __forceinline__ v16h load_frag_a(const _Float16* lds, int row,
                                            int stride, int lane) {
    const int half = (lane >> 4) & 1;
    const _Float16* p = lds + (size_t)row * stride + half * 8;
    FragU f;
    f.u[0] = *(const uint4*)(p);        // K = half*8 + 0..7
    f.u[1] = *(const uint4*)(p + 16);   // K = 16 + half*8 + 0..7
    return f.v;
}

// B-matrix fragment (32x16 f16, KxN) from an LDS tile stored [n][k]:
// lane n = lane&15 holds column n, K = {h*8 + 0..7, 16 + h*8 + 0..7}, h = lane>>4.
__device__ __forceinline__ v16h load_frag_b(const _Float16* lds_nk, int col_base,
                                            int stride, int lane) {
    const int n    = lane & 15;
    const int half = (lane >> 4) & 1;
    const _Float16* p = lds_nk + (size_t)(col_base + n) * stride + half * 8;
    FragU f;
    f.u[0] = *(const uint4*)(p);
    f.u[1] = *(const uint4*)(p + 16);
    return f.v;
}

// ---------------------------------------------------------------------------
// f32 -> f16 cast, 4 elements/thread
// ---------------------------------------------------------------------------
__global__ __launch_bounds__(256) void cast_f32_f16_kernel(
    const float* __restrict__ src, _Float16* __restrict__ dst, int n4) {
    int i = blockIdx.x * blockDim.x + threadIdx.x;
    if (i < n4) {
        float4 f = ((const float4*)src)[i];
        union { _Float16 h[4]; uint2 u; } r;
        r.h[0] = (_Float16)f.x; r.h[1] = (_Float16)f.y;
        r.h[2] = (_Float16)f.z; r.h[3] = (_Float16)f.w;
        ((uint2*)dst)[i] = r.u;
    }
}

// ---------------------------------------------------------------------------
// Y[M,N] = X[M,K] @ W[N,K]^T + bias   (torch Linear semantics)
// Block tile 128x128, BK=64, 256 threads = 8 waves, wave subtile 32x64.
// TDM-staged (or async-staged) double-buffered LDS tiles.
// ---------------------------------------------------------------------------
#define BM 128
#define BN 128
#define BK 64

template <typename OT>
__global__ __launch_bounds__(256) void gemm_bias_kernel(
    const _Float16* __restrict__ X,   // [M,K] row-major
    const _Float16* __restrict__ W,   // [N,K] row-major (Linear weight)
    const float*    __restrict__ bias,// [N]
    OT*             __restrict__ Y,   // [M,N]
    int M, int N, int K) {
    __shared__ _Float16 As[2 * BM * BK];  // [buf][m][k]  16KB per buffer
    __shared__ _Float16 Bs[2 * BN * BK];  // [buf][n][k]  (== W memory order)

    const int tid  = threadIdx.x;
    const int lane = tid & 31;
    const int wid  = tid >> 5;   // 0..7
    const int wm   = wid & 3;    // 4 wave-rows * 32 rows
    const int wn   = wid >> 2;   // 2 wave-cols * 64 cols
    const int m0   = blockIdx.y * BM;
    const int n0   = blockIdx.x * BN;

    const unsigned int asA = lds_addr_of(As);
    const unsigned int asB = lds_addr_of(Bs);

    v8f acc[2][4];
#pragma unroll
    for (int i = 0; i < 2; ++i)
#pragma unroll
        for (int j = 0; j < 4; ++j) acc[i][j] = zero8();

    // Stage one 128xBK A tile + 128xBK B tile into LDS buffer `buf`.
    auto issue_tiles = [&](int buf, int k0) {
#if HAVE_TDM
        if (wid == 0) {
            tdm_load_2d(asA + (unsigned int)(buf * BM * BK * 2),
                        X + (size_t)m0 * K + k0, BK, BM, (unsigned int)K);
            tdm_load_2d(asB + (unsigned int)(buf * BN * BK * 2),
                        W + (size_t)n0 * K + k0, BK, BN, (unsigned int)K);
        }
#else
        // 1024 16B chunks per tile, 256 threads -> 4 chunks per array each.
#pragma unroll
        for (int it = 0; it < 4; ++it) {
            int i = tid + it * 256;
            int r = i >> 3;            // 8 chunks per row
            int c = (i & 7) * 8;
            unsigned int l = (unsigned int)((buf * BM * BK + r * BK + c) * 2);
            async_load_b128(asA + l, X,
                            (unsigned int)(((size_t)(m0 + r) * K + k0 + c) * 2));
            async_load_b128(asB + l, W,
                            (unsigned int)(((size_t)(n0 + r) * K + k0 + c) * 2));
        }
#endif
    };

    auto wait_stage = [&]() {
#if HAVE_TDM
        if (wid == 0) __builtin_amdgcn_s_wait_tensorcnt(0);
#else
        wait_async0();
#endif
    };

    issue_tiles(0, 0);

    int buf = 0;
    for (int k0 = 0; k0 < K; k0 += BK, buf ^= 1) {
        wait_stage();         // staging of current buffer complete (issuer)
        __syncthreads();      // published to all waves; prev buffer free
        if (k0 + BK < K) issue_tiles(buf ^ 1, k0 + BK);

        const _Float16* Ac = As + buf * BM * BK;
        const _Float16* Bc = Bs + buf * BN * BK;

#pragma unroll
        for (int c = 0; c < 2; ++c) {   // two K=32 sub-steps within BK=64
            v16h afr[2], bfr[4];
#pragma unroll
            for (int i = 0; i < 2; ++i)
                afr[i] = load_frag_a(Ac + c * 32,
                                     wm * 32 + i * 16 + (lane & 15), BK, lane);
#pragma unroll
            for (int j = 0; j < 4; ++j)
                bfr[j] = load_frag_b(Bc + c * 32, wn * 64 + j * 16, BK, lane);

#pragma unroll
            for (int i = 0; i < 2; ++i)
#pragma unroll
                for (int j = 0; j < 4; ++j)
                    acc[i][j] = __builtin_amdgcn_wmma_f32_16x16x32_f16(
                        false, afr[i], false, bfr[j], (short)0, acc[i][j],
                        false, false);
        }
    }

    // Writeback. C layout: VGPR e -> row e (lanes<16) / e+8 (lanes>=16),
    // col = lane&15.
    const int lhalf = lane >> 4;
    const int ncol  = lane & 15;
#pragma unroll
    for (int i = 0; i < 2; ++i) {
#pragma unroll
        for (int j = 0; j < 4; ++j) {
            int col = n0 + wn * 64 + j * 16 + ncol;
            float bv = bias[col];
#pragma unroll
            for (int e = 0; e < 8; ++e) {
                int row = m0 + wm * 32 + i * 16 + e + 8 * lhalf;
                Y[(size_t)row * N + col] = (OT)(acc[i][j][e] + bv);
            }
        }
    }
}

// ---------------------------------------------------------------------------
// Flash attention per (b, h): 64-row Q tile per block, online softmax.
// 128 threads = 4 waves; each wave owns 16 query rows.
// Q/K tiles staged via async-to-LDS; V staged transposed manually.
// ---------------------------------------------------------------------------
__global__ __launch_bounds__(128) void flash_attn_kernel(
    const _Float16* __restrict__ Qh,  // [B*S, D], head h at col h*64
    const _Float16* __restrict__ Kh,
    const _Float16* __restrict__ Vh,
    const int*      __restrict__ mask, // [B,S,S]
    _Float16*       __restrict__ ctx,  // [B*S, D]
    int S, int D) {
    __shared__ _Float16 Qs[64 * 64];  // [m][dk]
    __shared__ _Float16 Ks[64 * 64];  // [n][dk]  (B operand of Q*K^T)
    __shared__ _Float16 Vt[64 * 64];  // [dk][key] (B operand of P*V)
    __shared__ _Float16 Ps[64 * 64];  // [m][key]  (A operand of P*V)

    const int tid  = threadIdx.x;
    const int lane = tid & 31;
    const int wid  = tid >> 5;        // 0..3
    const int m0   = blockIdx.x * 64;
    const int h    = blockIdx.y;
    const int b    = blockIdx.z;
    const size_t base = (size_t)b * S * D + (size_t)h * 64;

    const unsigned int qsA = lds_addr_of(Qs);
    const unsigned int ksA = lds_addr_of(Ks);

    // Stage Q tile 64x64 (async)
#pragma unroll
    for (int it = 0; it < 4; ++it) {
        int i = tid + it * 128;
        int r = i >> 3;               // 8 chunks of 8 f16 per row
        int c = (i & 7) * 8;
        async_load_b128(qsA + (unsigned int)((r * 64 + c) * 2), Qh,
                        (unsigned int)((base + (size_t)(m0 + r) * D + c) * 2));
    }
    wait_async0();
    __syncthreads();

    v16h qf[2];
#pragma unroll
    for (int c = 0; c < 2; ++c)
        qf[c] = load_frag_a(Qs + c * 32, wid * 16 + (lane & 15), 64, lane);

    const int lhalf = lane >> 4;
    const int ncol  = lane & 15;

    float Mr[8], Lr[8];
    v8f o[4];
#pragma unroll
    for (int e = 0; e < 8; ++e) { Mr[e] = -3.0e38f; Lr[e] = 0.f; }
#pragma unroll
    for (int ns = 0; ns < 4; ++ns) o[ns] = zero8();

    const float scale = 0.125f;  // 1/sqrt(DK=64)

    for (int n0 = 0; n0 < S; n0 += 64) {
        // Stage K tile [n][dk] async; V tile transposed [dk][key] manually.
#pragma unroll
        for (int it = 0; it < 4; ++it) {
            int i = tid + it * 128;
            int r = i >> 3;
            int c = (i & 7) * 8;
            async_load_b128(ksA + (unsigned int)((r * 64 + c) * 2), Kh,
                            (unsigned int)((base + (size_t)(n0 + r) * D + c) * 2));
            union { uint4 u; _Float16 hh[8]; } t;
            t.u = *(const uint4*)(&Vh[base + (size_t)(n0 + r) * D + c]);
#pragma unroll
            for (int e = 0; e < 8; ++e) Vt[(c + e) * 64 + r] = t.hh[e];
        }
        wait_async0();
        __syncthreads();

        // Scores S = Q*K^T for this wave's 16 rows x 64 keys.
        v8f sacc[4];
#pragma unroll
        for (int ns = 0; ns < 4; ++ns) {
            sacc[ns] = zero8();
#pragma unroll
            for (int c = 0; c < 2; ++c) {
                v16h bf = load_frag_b(Ks + c * 32, ns * 16, 64, lane);
                sacc[ns] = __builtin_amdgcn_wmma_f32_16x16x32_f16(
                    false, qf[c], false, bf, (short)0, sacc[ns], false, false);
            }
        }

        // Scale + mask.
#pragma unroll
        for (int ns = 0; ns < 4; ++ns) {
#pragma unroll
            for (int e = 0; e < 8; ++e) {
                int gr = m0 + wid * 16 + e + 8 * lhalf;
                int gc = n0 + ns * 16 + ncol;
                float s = sacc[ns][e] * scale;
                int mk = mask[((size_t)b * S + gr) * S + gc];
                sacc[ns][e] = (mk == 0) ? -1e9f : s;
            }
        }

        // Online softmax update (reduce over 4 n-subtiles + 16 lanes).
        float nmax[8];
#pragma unroll
        for (int e = 0; e < 8; ++e) {
            float m = fmaxf(fmaxf(sacc[0][e], sacc[1][e]),
                            fmaxf(sacc[2][e], sacc[3][e]));
#pragma unroll
            for (int off = 1; off < 16; off <<= 1)
                m = fmaxf(m, __shfl_xor(m, off, 32));
            nmax[e] = fmaxf(Mr[e], m);
        }
#pragma unroll
        for (int e = 0; e < 8; ++e) {
            float alpha = __expf(Mr[e] - nmax[e]);
            Lr[e] *= alpha;
#pragma unroll
            for (int ns = 0; ns < 4; ++ns) o[ns][e] *= alpha;
            float rs = 0.f;
#pragma unroll
            for (int ns = 0; ns < 4; ++ns) {
                float p = __expf(sacc[ns][e] - nmax[e]);
                sacc[ns][e] = p;
                rs += p;
            }
#pragma unroll
            for (int off = 1; off < 16; off <<= 1)
                rs += __shfl_xor(rs, off, 32);
            Lr[e] += rs;
            Mr[e] = nmax[e];
        }

        // P -> LDS (f16, [m][key]) so it can be re-read in A-fragment layout.
#pragma unroll
        for (int ns = 0; ns < 4; ++ns)
#pragma unroll
            for (int e = 0; e < 8; ++e) {
                int rl = wid * 16 + e + 8 * lhalf;
                int cl = ns * 16 + ncol;
                Ps[rl * 64 + cl] = (_Float16)sacc[ns][e];
            }

        // O += P * V
        v16h pf[2];
#pragma unroll
        for (int c = 0; c < 2; ++c)
            pf[c] = load_frag_a(Ps + c * 32, wid * 16 + (lane & 15), 64, lane);
#pragma unroll
        for (int ns = 0; ns < 4; ++ns) {
#pragma unroll
            for (int c = 0; c < 2; ++c) {
                v16h vb = load_frag_b(Vt + c * 32, ns * 16, 64, lane);
                o[ns] = __builtin_amdgcn_wmma_f32_16x16x32_f16(
                    false, pf[c], false, vb, (short)0, o[ns], false, false);
            }
        }
        __syncthreads();   // protect Ks/Vt/Ps before restage
    }

    // Normalize and write ctx (f16).
#pragma unroll
    for (int ns = 0; ns < 4; ++ns)
#pragma unroll
        for (int e = 0; e < 8; ++e) {
            int row = m0 + wid * 16 + e + 8 * lhalf;
            int col = ns * 16 + ncol;
            float val = o[ns][e] / Lr[e];
            ctx[base + (size_t)row * D + col] = (_Float16)val;
        }
}

// ---------------------------------------------------------------------------
// Launch
// ---------------------------------------------------------------------------
extern "C" void kernel_launch(void* const* d_in, const int* in_sizes, int n_in,
                              void* d_out, int out_size, void* d_ws, size_t ws_size,
                              hipStream_t stream) {
    (void)in_sizes; (void)n_in; (void)out_size; (void)ws_size;
    const int B = 2, S = 2048, D = 1024, H = 16;
    const int M = B * S;                 // 4096
    const size_t MD = (size_t)M * D;     // 4,194,304
    const size_t DD = (size_t)D * D;     // 1,048,576

    const float* q    = (const float*)d_in[0];
    const float* k    = (const float*)d_in[1];
    const float* v    = (const float*)d_in[2];
    const int*   mask = (const int*)  d_in[3];
    const float* Wq   = (const float*)d_in[4];
    const float* bq   = (const float*)d_in[5];
    const float* Wk   = (const float*)d_in[6];
    const float* bk   = (const float*)d_in[7];
    const float* Wv   = (const float*)d_in[8];
    const float* bv   = (const float*)d_in[9];
    const float* Wo   = (const float*)d_in[10];
    const float* bo   = (const float*)d_in[11];
    float* out = (float*)d_out;

    // Workspace (f16): 4 activation buffers (reused) + 4 weight buffers.
    _Float16* ws   = (_Float16*)d_ws;
    _Float16* buf0 = ws;            // q16 -> later Kh
    _Float16* buf1 = ws + MD;       // k16 -> later Vh
    _Float16* buf2 = ws + 2 * MD;   // v16 -> later ctx
    _Float16* buf3 = ws + 3 * MD;   // Qh
    _Float16* wq16 = ws + 4 * MD;
    _Float16* wk16 = wq16 + DD;
    _Float16* wv16 = wk16 + DD;
    _Float16* wo16 = wv16 + DD;

    // Casts
    {
        int n4 = (int)(MD / 4);
        int blocks = (n4 + 255) / 256;
        cast_f32_f16_kernel<<<blocks, 256, 0, stream>>>(q, buf0, n4);
        cast_f32_f16_kernel<<<blocks, 256, 0, stream>>>(k, buf1, n4);
        cast_f32_f16_kernel<<<blocks, 256, 0, stream>>>(v, buf2, n4);
        int w4 = (int)(DD / 4);
        int wblocks = (w4 + 255) / 256;
        cast_f32_f16_kernel<<<wblocks, 256, 0, stream>>>(Wq, wq16, w4);
        cast_f32_f16_kernel<<<wblocks, 256, 0, stream>>>(Wk, wk16, w4);
        cast_f32_f16_kernel<<<wblocks, 256, 0, stream>>>(Wv, wv16, w4);
        cast_f32_f16_kernel<<<wblocks, 256, 0, stream>>>(Wo, wo16, w4);
    }

    dim3 gg(D / BN, M / BM);  // (8, 32)
    // Q/K/V projections (note buffer reuse ordering).
    gemm_bias_kernel<_Float16><<<gg, 256, 0, stream>>>(buf0, wq16, bq, buf3, M, D, D); // Qh=buf3
    gemm_bias_kernel<_Float16><<<gg, 256, 0, stream>>>(buf1, wk16, bk, buf0, M, D, D); // Kh=buf0
    gemm_bias_kernel<_Float16><<<gg, 256, 0, stream>>>(buf2, wv16, bv, buf1, M, D, D); // Vh=buf1

    dim3 fg(S / 64, H, B);    // (32, 16, 2)
    flash_attn_kernel<<<fg, 128, 0, stream>>>(buf3, buf0, buf1, mask, buf2, S, D); // ctx=buf2

    // Output projection -> fp32 out.
    gemm_bias_kernel<float><<<gg, 256, 0, stream>>>(buf2, wo16, bo, out, M, D, D);
}
